// MHSA_35716948033631
// MI455X (gfx1250) — matrix operand order repaired and checked by reading, not compiled, not run on hardware
//
#include <hip/hip_runtime.h>
#include <hip/hip_bf16.h>

// ---------------------------------------------------------------------------
// MHSA with relative-position-bias MLP, CDNA5 (gfx1250) WMMA bf16 pipeline.
//   P=2048, D=1024, H=16, DH=64, HID=32
// ---------------------------------------------------------------------------

constexpr int PP  = 2048;
constexpr int DD  = 1024;
constexpr int HH  = 16;
constexpr int DHD = 64;       // head dim
constexpr int HID = 32;

typedef __attribute__((ext_vector_type(16))) __bf16 v16bf;
typedef __attribute__((ext_vector_type(8)))  float  v8f;

union Bf4 { __bf16 h[4]; uint2 u; };

// pack 4 floats -> 4 bf16 -> one 8-byte LDS store
__device__ inline void store_bf16x4(__bf16* dst, float a, float b, float c, float d) {
  Bf4 t;
  t.h[0] = (__bf16)a; t.h[1] = (__bf16)b; t.h[2] = (__bf16)c; t.h[3] = (__bf16)d;
  *(uint2*)dst = t.u;
}

// ---------------------------------------------------------------------------
// WMMA fragment loaders (wave32; k order contiguous, not ISA 7.12.2 table order). All loads contiguous.
// A 16x32 bf16 (MxK), row-major tile:
//   lanes 0-15  : row M=lane,    regs hold K = 0..15
//   lanes 16-31 : row M=lane-16, regs hold K = 16..31
__device__ inline v16bf load_a_frag(const __bf16* base, int ld, int lane) {
  const int row = lane & 15;
  const int kh  = (lane >> 4) & 1;
  const __bf16* p0 = base + row * ld + kh * 16;
  const __bf16* p1 = p0 + 8;
  v16bf a;
#pragma unroll
  for (int i = 0; i < 8; ++i) { a[i] = p0[i]; a[8 + i] = p1[i]; }
  return a;
}

// B 32x16 bf16 (KxN) from a TRANSPOSED tile: element (k,n) = tileT[n*ld + k].
//   lanes 0-15 : col N=lane, K=0..15 ; lanes 16-31 : col N=lane-16, K=16..31
// -> 16 contiguous bf16 per lane (32B).
__device__ inline v16bf load_bT_frag(const __bf16* tileT, int ld, int lane) {
  const int col = lane & 15;
  const int kh  = (lane >> 4) & 1;
  const __bf16* p = tileT + col * ld + kh * 16;
  v16bf b;
#pragma unroll
  for (int i = 0; i < 16; ++i) b[i] = p[i];
  return b;
}

// ---------------------------------------------------------------------------
// Kernel 1/4: generic C = A(f32) x B(f32) GEMM via bf16 WMMA, f32 accumulate.
// Block tile 128x128, K-step 32, 256 threads (8 waves, 2x4 16x16 tiles each).
// B is staged TRANSPOSED in LDS for contiguous fragment loads.
// OUT_BF16=1: write bf16, scaling columns < scaleN by scaleVal (q-scale fold).
// ---------------------------------------------------------------------------
template <int OUT_BF16>
__global__ __launch_bounds__(256) void gemm_wmma_bf16(
    const float* __restrict__ A, const float* __restrict__ B,
    float* __restrict__ Cf, __bf16* __restrict__ Cb,
    int M, int N, int K, int scaleN, float scaleVal)
{
  constexpr int LDA = 40;   // 32 + 8 pad (bf16); 80B rows, 8B-aligned chunks
  constexpr int LDBT = 40;  // transposed B rows: 32 K + 8 pad
  __shared__ __bf16 As[128 * LDA];
  __shared__ __bf16 BsT[128 * LDBT];   // BsT[n][k]

  const int tid  = threadIdx.x;
  const int lane = tid & 31;
  const int w    = tid >> 5;
  const int m0   = blockIdx.y * 128;
  const int n0   = blockIdx.x * 128;
  const int wm   = (w & 3) * 32;   // wave m offset in block
  const int wn   = (w >> 2) * 64;  // wave n offset in block

  v8f acc[2][4] = {};

  for (int kt = 0; kt < K; kt += 32) {
    __syncthreads();
    // stage A tile 128x32 (f32 -> bf16), row-major
#pragma unroll
    for (int it = 0; it < 4; ++it) {
      const int r = (tid >> 3) + it * 32;
      const int c = (tid & 7) * 4;
      const float4 f = *(const float4*)(A + (size_t)(m0 + r) * K + kt + c);
      store_bf16x4(&As[r * LDA + c], f.x, f.y, f.z, f.w);
    }
    // stage B tile 32x128 TRANSPOSED: lanes consecutive in n -> coalesced loads
#pragma unroll
    for (int it = 0; it < 4; ++it) {
      const int slot = tid + it * 256;
      const int n  = slot & 127;
      const int kg = slot >> 7;                 // 0..7 (groups of 4 k)
      const float* src = B + (size_t)(kt + kg * 4) * N + n0 + n;
      const float f0 = src[0];
      const float f1 = src[(size_t)N];
      const float f2 = src[2 * (size_t)N];
      const float f3 = src[3 * (size_t)N];
      store_bf16x4(&BsT[n * LDBT + kg * 4], f0, f1, f2, f3);
    }
    __syncthreads();

    v16bf a[2];
#pragma unroll
    for (int mi = 0; mi < 2; ++mi)
      a[mi] = load_a_frag(&As[(wm + mi * 16) * LDA], LDA, lane);
#pragma unroll
    for (int ni = 0; ni < 4; ++ni) {
      v16bf b = load_bT_frag(&BsT[(wn + ni * 16) * LDBT], LDBT, lane);
#pragma unroll
      for (int mi = 0; mi < 2; ++mi)
        acc[mi][ni] = __builtin_amdgcn_wmma_f32_16x16x32_bf16(
            false, a[mi], false, b, (short)0, acc[mi][ni], false, false);
    }
  }

  // epilogue: C/D layout — reg r holds M=r (lanes 0-15) / M=r+8 (lanes 16-31)
  const int ncol = lane & 15;
  const int mhi  = (lane >> 4) ? 8 : 0;
#pragma unroll
  for (int mi = 0; mi < 2; ++mi)
#pragma unroll
    for (int ni = 0; ni < 4; ++ni)
#pragma unroll
      for (int r = 0; r < 8; ++r) {
        const int m = m0 + wm + mi * 16 + mhi + r;
        const int n = n0 + wn + ni * 16 + ncol;
        float v = acc[mi][ni][r];
        if (OUT_BF16) {
          if (n < scaleN) v *= scaleVal;
          Cb[(size_t)m * N + n] = (__bf16)v;
        } else {
          Cf[(size_t)m * N + n] = v;
        }
      }
}

// ---------------------------------------------------------------------------
// Kernel 2: relative-position-bias MLP: (dx,dy) -> 32 -> 32 -> 1, ReLU.
// ---------------------------------------------------------------------------
__global__ __launch_bounds__(256) void rpb_mlp_kernel(
    const float* __restrict__ rel,
    const float* __restrict__ w1, const float* __restrict__ b1,
    const float* __restrict__ w2, const float* __restrict__ b2,
    const float* __restrict__ w3, const float* __restrict__ b3,
    float* __restrict__ bias)
{
  __shared__ float sw1[2 * HID], sb1[HID], sw2[HID * HID], sb2[HID], sw3[HID], sb3;
  for (int i = threadIdx.x; i < 2 * HID; i += 256) sw1[i] = w1[i];
  for (int i = threadIdx.x; i < HID; i += 256) { sb1[i] = b1[i]; sb2[i] = b2[i]; sw3[i] = w3[i]; }
  for (int i = threadIdx.x; i < HID * HID; i += 256) sw2[i] = w2[i];
  if (threadIdx.x == 0) sb3 = b3[0];
  __syncthreads();

  const size_t idx = (size_t)blockIdx.x * 256 + threadIdx.x;
  if (idx >= (size_t)PP * PP) return;
  const float dx = rel[idx * 2 + 0];
  const float dy = rel[idx * 2 + 1];

  float h1[HID];
#pragma unroll
  for (int k = 0; k < HID; ++k)
    h1[k] = fmaxf(fmaf(dx, sw1[k], fmaf(dy, sw1[HID + k], sb1[k])), 0.0f);

  float o = sb3;
#pragma unroll 4
  for (int k = 0; k < HID; ++k) {
    float acc = sb2[k];
#pragma unroll
    for (int j = 0; j < HID; ++j) acc = fmaf(h1[j], sw2[j * HID + k], acc);
    o = fmaf(fmaxf(acc, 0.0f), sw3[k], o);
  }
  bias[idx] = o;
}

// ---------------------------------------------------------------------------
// Kernel 3: flash attention per (head, 64-row block); BN=64 column blocks.
// qkv is bf16 [P][3D] with q pre-scaled by 1/sqrt(DH). bias is f32 [P][P].
// V is staged TRANSPOSED (VsT[d][key]) via wide loads + 4x4 register
// transpose so PV B-fragments load contiguously.
// ---------------------------------------------------------------------------
__global__ __launch_bounds__(256) void attn_flash_kernel(
    const __bf16* __restrict__ qkv, const float* __restrict__ bias,
    float* __restrict__ out)
{
  constexpr int BM = 64, BN = 64;
  constexpr int DHp = 72;   // bf16 row pad: 144B rows -> 16B-aligned chunks
  constexpr int SP  = 72;   // f32 row pad: 288B rows -> float4-aligned
  __shared__ __bf16 Qs [BM * DHp];    // Qs[m][d]
  __shared__ __bf16 Ks [BN * DHp];    // Ks[key][d]   (S: B^T source)
  __shared__ __bf16 VsT[DHD * DHp];   // VsT[d][key]  (PV: B^T source)
  __shared__ __bf16 Ps [BM * DHp];    // probs bf16
  __shared__ float  Ss [BM * SP];     // bias tile, then scores
  __shared__ float  alpha_s[BM];
  __shared__ float  linv_s[BM];

  const int tid  = threadIdx.x;
  const int lane = tid & 31;
  const int w    = tid >> 5;        // wave 0..7
  const int mb   = blockIdx.x;      // row block
  const int h    = blockIdx.y;      // head
  const int row0 = mb * BM;
  const int qcol = h * DHD;
  const int kcol = DD + h * DHD;
  const int vcol = 2 * DD + h * DHD;

  // wave tile assignment: 4x4 grid of 16x16 tiles; wave -> (tm, 2 n-tiles)
  const int tm  = w & 3;
  const int tnb = (w >> 2) * 2;

  // softmax ownership: 4 threads per row
  const int sr = tid >> 2;
  const int sq = tid & 3;
  float m_i = -3.0e38f, l_i = 0.0f;

  // load Q tile (pre-scaled bf16), 4 bf16 per slot
  for (int i = tid; i < BM * DHD / 4; i += 256) {
    const int r = i >> 4, c4 = (i & 15) * 4;
    *(uint2*)&Qs[r * DHp + c4] =
        *(const uint2*)(qkv + (size_t)(row0 + r) * (3 * DD) + qcol + c4);
  }

  v8f o_acc[2] = {};
  const int ncol = lane & 15;
  const int mhi  = (lane >> 4) ? 8 : 0;

  for (int nb = 0; nb < PP / BN; ++nb) {
    __syncthreads();  // protect Ks/VsT/Ss reuse from previous iteration
    const int krow0 = nb * BN;
    // K tile row-major, 4 bf16 per slot (wide, coalesced)
#pragma unroll
    for (int it = 0; it < 4; ++it) {
      const int slot = tid + it * 256;
      const int r = slot >> 4, c4 = (slot & 15) * 4;
      *(uint2*)&Ks[r * DHp + c4] =
          *(const uint2*)(qkv + (size_t)(krow0 + r) * (3 * DD) + kcol + c4);
    }
    // V tile: each thread owns a 4(key)x4(d) micro-block; wide global loads
    // (lanes consecutive in d -> coalesced), 4x4 bf16 register transpose,
    // wide LDS stores into VsT[d][key].
    {
      const int kb = tid >> 4;        // key block 0..15
      const int db = tid & 15;        // d block 0..15
      const int k4 = kb * 4, d4 = db * 4;
      const __bf16* src = qkv + (size_t)(krow0 + k4) * (3 * DD) + vcol + d4;
      Bf4 in[4];
#pragma unroll
      for (int j = 0; j < 4; ++j)
        in[j].u = *(const uint2*)(src + (size_t)j * (3 * DD));
#pragma unroll
      for (int i = 0; i < 4; ++i) {
        Bf4 o;
#pragma unroll
        for (int j = 0; j < 4; ++j) o.h[j] = in[j].h[i];
        *(uint2*)&VsT[(d4 + i) * DHp + k4] = o.u;
      }
    }
    // bias tile -> Ss (float4)
    for (int i = tid; i < BM * BN / 4; i += 256) {
      const int r = i >> 4, c4 = (i & 15) * 4;
      *(float4*)&Ss[r * SP + c4] =
          *(const float4*)(bias + (size_t)(row0 + r) * PP + krow0 + c4);
    }
    __syncthreads();

    // S = Q * K^T (+ bias already sitting in Ss)
    v8f s_acc[2] = {};
#pragma unroll
    for (int kk = 0; kk < DHD; kk += 32) {
      v16bf a = load_a_frag(&Qs[(tm * 16) * DHp + kk], DHp, lane);
#pragma unroll
      for (int ni = 0; ni < 2; ++ni) {
        v16bf b = load_bT_frag(&Ks[((tnb + ni) * 16) * DHp + kk], DHp, lane);
        s_acc[ni] = __builtin_amdgcn_wmma_f32_16x16x32_bf16(
            false, a, false, b, (short)0, s_acc[ni], false, false);
      }
    }
#pragma unroll
    for (int ni = 0; ni < 2; ++ni)
#pragma unroll
      for (int r = 0; r < 8; ++r) {
        const int m = tm * 16 + mhi + r;
        const int n = (tnb + ni) * 16 + ncol;
        Ss[m * SP + n] += s_acc[ni][r];   // each (m,n) owned by one lane
      }
    __syncthreads();

    // online softmax over this 64-wide block (4 lanes per row)
    {
      const float4* srow4 = (const float4*)&Ss[sr * SP + sq * 16];
      float sv[16];
#pragma unroll
      for (int g = 0; g < 4; ++g) {
        const float4 f = srow4[g];
        sv[g * 4 + 0] = f.x; sv[g * 4 + 1] = f.y;
        sv[g * 4 + 2] = f.z; sv[g * 4 + 3] = f.w;
      }
      float lmax = -3.0e38f;
#pragma unroll
      for (int c = 0; c < 16; ++c) lmax = fmaxf(lmax, sv[c]);
      lmax = fmaxf(lmax, __shfl_xor(lmax, 1, 32));
      lmax = fmaxf(lmax, __shfl_xor(lmax, 2, 32));
      const float m_new = fmaxf(m_i, lmax);
      float psum = 0.0f;
      float pv[16];
#pragma unroll
      for (int c = 0; c < 16; ++c) {
        pv[c] = __expf(sv[c] - m_new);
        psum += pv[c];
      }
      __bf16* prow = &Ps[sr * DHp + sq * 16];
#pragma unroll
      for (int c4 = 0; c4 < 16; c4 += 4)
        store_bf16x4(&prow[c4], pv[c4], pv[c4 + 1], pv[c4 + 2], pv[c4 + 3]);
      psum += __shfl_xor(psum, 1, 32);
      psum += __shfl_xor(psum, 2, 32);
      const float alpha = __expf(m_i - m_new);
      l_i = l_i * alpha + psum;
      m_i = m_new;
      if (sq == 0) alpha_s[sr] = alpha;
    }
    __syncthreads();

    // O = alpha*O + P * V
#pragma unroll
    for (int ni = 0; ni < 2; ++ni)
#pragma unroll
      for (int r = 0; r < 8; ++r)
        o_acc[ni][r] *= alpha_s[tm * 16 + mhi + r];
#pragma unroll
    for (int kk = 0; kk < BN; kk += 32) {
      v16bf a = load_a_frag(&Ps[(tm * 16) * DHp + kk], DHp, lane);
#pragma unroll
      for (int ni = 0; ni < 2; ++ni) {
        // B[k][d] = V[key=k][d] = VsT[d][k]; d-tile base, K offset kk
        v16bf b = load_bT_frag(&VsT[((tnb + ni) * 16) * DHp + kk], DHp, lane);
        o_acc[ni] = __builtin_amdgcn_wmma_f32_16x16x32_bf16(
            false, a, false, b, (short)0, o_acc[ni], false, false);
      }
    }
  }

  if (sq == 0) linv_s[sr] = 1.0f / l_i;
  __syncthreads();

  // write normalized O to [P][D] at head column block
#pragma unroll
  for (int ni = 0; ni < 2; ++ni)
#pragma unroll
    for (int r = 0; r < 8; ++r) {
      const int m = tm * 16 + mhi + r;
      const int d = (tnb + ni) * 16 + ncol;
      out[(size_t)(row0 + m) * DD + h * DHD + d] = o_acc[ni][r] * linv_s[m];
    }
}

// ---------------------------------------------------------------------------
// Host launcher
// ---------------------------------------------------------------------------
extern "C" void kernel_launch(void* const* d_in, const int* in_sizes, int n_in,
                              void* d_out, int out_size, void* d_ws, size_t ws_size,
                              hipStream_t stream) {
  const float* x      = (const float*)d_in[0];
  const float* rel    = (const float*)d_in[1];
  const float* w_qkv  = (const float*)d_in[2];
  const float* w_proj = (const float*)d_in[3];
  const float* rpb_w1 = (const float*)d_in[4];
  const float* rpb_b1 = (const float*)d_in[5];
  const float* rpb_w2 = (const float*)d_in[6];
  const float* rpb_b2 = (const float*)d_in[7];
  const float* rpb_w3 = (const float*)d_in[8];
  const float* rpb_b3 = (const float*)d_in[9];
  float* out = (float*)d_out;

  char* ws = (char*)d_ws;
  __bf16* qkv_bf = (__bf16*)ws;                                      // 2048*3072 bf16
  float*  bias   = (float*)(ws + (size_t)PP * 3 * DD * 2);           // 2048*2048 f32
  float*  attn   = (float*)(ws + (size_t)PP * 3 * DD * 2
                               + (size_t)PP * PP * 4);               // 2048*1024 f32

  // 1) qkv = x @ w_qkv  (bf16 out, q columns scaled by 1/sqrt(DH)=0.125)
  dim3 g1(3 * DD / 128, PP / 128);
  gemm_wmma_bf16<1><<<g1, 256, 0, stream>>>(x, w_qkv, nullptr, qkv_bf,
                                            PP, 3 * DD, DD, DD, 0.125f);

  // 2) bias = MLP(rel)
  rpb_mlp_kernel<<<(PP * PP) / 256, 256, 0, stream>>>(
      rel, rpb_w1, rpb_b1, rpb_w2, rpb_b2, rpb_w3, rpb_b3, bias);

  // 3) flash attention per (row-block, head)
  dim3 g3(PP / 64, HH);
  attn_flash_kernel<<<g3, 256, 0, stream>>>(qkv_bf, bias, attn);

  // 4) out = attn @ w_proj (f32 out)
  dim3 g4(DD / 128, PP / 128);
  gemm_wmma_bf16<0><<<g4, 256, 0, stream>>>(attn, w_proj, out, nullptr,
                                            PP, DD, DD, 0, 1.0f);
}
